// transformer_block_81655918232007
// MI455X (gfx1250) — matrix operand order repaired and checked by reading, not compiled, not run on hardware
//
#include <hip/hip_runtime.h>
#include <math.h>
#include <stdint.h>

typedef unsigned short u16;
typedef __attribute__((ext_vector_type(16))) __bf16 v16bf;
typedef __attribute__((ext_vector_type(8)))  float  v8f;
typedef __attribute__((ext_vector_type(4)))  unsigned int v4u;
typedef __attribute__((ext_vector_type(4)))  int v4i;

#define D_MODEL 1024
#define NHEADS  16
#define DK      64
#define D_FF    4096
#define SEQ     2048
#define BATCH   4
#define MROWS   (BATCH*SEQ)   // 8192

// ---- CDNA5 async global->LDS path (guarded; falls back to load+ds_store) ----
#if defined(__gfx1250__) && __has_builtin(__builtin_amdgcn_global_load_async_to_lds_b128)
#define HAS_ASYNC_LDS 1
#else
#define HAS_ASYNC_LDS 0
#endif

#if __has_builtin(__builtin_amdgcn_s_wait_asynccnt)
#define WAIT_ASYNC(n) __builtin_amdgcn_s_wait_asynccnt(n)
#else
#define WAIT_ASYNC(n) asm volatile("s_wait_asynccnt %0" :: "i"(n) : "memory")
#endif

// builtin signature (probed via compile diagnostics): param0 = AS1 v4i*, param1 = AS3 v4i*, imm, imm
#define AS1V(p) ((__attribute__((address_space(1))) v4i*)(unsigned long long)(uintptr_t)(p))
#define AS3V(p) ((__attribute__((address_space(3))) v4i*)(unsigned)(uintptr_t)(p))

union Frag {
    v4u   q[2];     // 2 x 16B
    v16bf v;        // WMMA operand (16 bf16 = 8 VGPRs)
    u16   s[16];
};

__device__ __forceinline__ u16 f2bf(float x) {
    unsigned u = __float_as_uint(x);
    u += 0x7FFFu + ((u >> 16) & 1u);          // round-to-nearest-even
    return (u16)(u >> 16);
}
__device__ __forceinline__ float bf2f(u16 h) {
    return __uint_as_float(((unsigned)h) << 16);
}
__device__ __forceinline__ v8f vzero8() {
    v8f z;
#pragma unroll
    for (int i = 0; i < 8; ++i) z[i] = 0.0f;
    return z;
}

// ---------------------------------------------------------------- fp32->bf16
__global__ void f2bf_kernel(const float* __restrict__ src, u16* __restrict__ dst, int n) {
    int i = blockIdx.x * 256 + threadIdx.x;
    if (i < n) dst[i] = f2bf(src[i]);
}

// ---------------------------------------------------------------- RMSNorm -> bf16
__global__ __launch_bounds__(256) void rmsnorm_kernel(const float* __restrict__ X,
                                                      const float* __restrict__ g,
                                                      u16* __restrict__ out) {
    int row = blockIdx.x, tid = threadIdx.x;
    const float* xr = X + (size_t)row * D_MODEL;
    float xv[4], ss = 0.f;
#pragma unroll
    for (int i = 0; i < 4; ++i) { xv[i] = xr[tid + i * 256]; ss += xv[i] * xv[i]; }
    __shared__ float red[256];
    red[tid] = ss; __syncthreads();
    for (int s = 128; s > 0; s >>= 1) {
        if (tid < s) red[tid] += red[tid + s];
        __syncthreads();
    }
    float rinv = rsqrtf(red[0] * (1.0f / D_MODEL) + 1e-5f);
#pragma unroll
    for (int i = 0; i < 4; ++i)
        out[(size_t)row * D_MODEL + tid + i * 256] = f2bf(xv[i] * g[tid + i * 256] * rinv);
}

// ---------------------------------------------------------------- WMMA GEMM  C = A(MxK) * W(NxK)^T
// LDS-staged, double-buffered, K-slice = 64.
// modes: 0 = f32 store, 1 = f32 + residual R, 2 = bf16 store, 3 = bf16 silu(H1)*acc
__global__ __launch_bounds__(256) void gemm_kernel(const u16* __restrict__ A,
                                                   const u16* __restrict__ W,
                                                   int N, int K, int mode,
                                                   float* __restrict__ Cf,
                                                   const float* __restrict__ R,
                                                   u16* __restrict__ Cb,
                                                   const u16* __restrict__ Hb) {
    const int tid  = threadIdx.x;
    const int wave = tid >> 5;
    const int lane = tid & 31;
    const int g    = lane >> 4;
    const int ln   = lane & 15;
    const int mBase = blockIdx.y * 128;
    const int nBase = blockIdx.x * 128;

    // double-buffered tiles: A[128][64], B[128][64] bf16  (16KB each, 64KB total)
    __shared__ __attribute__((aligned(16))) u16 sA[2][128 * 64];
    __shared__ __attribute__((aligned(16))) u16 sB[2][128 * 64];

    const int srow = tid >> 1;          // 0..127
    const int scol = (tid & 1) * 32;    // halves within 64-wide slice

    v8f acc[4][2];
#pragma unroll
    for (int rt = 0; rt < 4; ++rt)
#pragma unroll
        for (int ct = 0; ct < 2; ++ct) acc[rt][ct] = vzero8();

    const int nks = K >> 6;             // K / 64

    // -------- tile stager: each thread moves 64B of A and 64B of B --------
    auto stage = [&](int buf, int kb) {
        const u16* ga = A + (size_t)(mBase + srow) * K + kb + scol;
        const u16* gb = W + (size_t)(nBase + srow) * K + kb + scol;
        u16* la = &sA[buf][srow * 64 + scol];
        u16* lb = &sB[buf][srow * 64 + scol];
#if HAS_ASYNC_LDS
#pragma unroll
        for (int j = 0; j < 4; ++j)
            __builtin_amdgcn_global_load_async_to_lds_b128(AS1V(ga + j * 8), AS3V(la + j * 8), 0, 0);
#pragma unroll
        for (int j = 0; j < 4; ++j)
            __builtin_amdgcn_global_load_async_to_lds_b128(AS1V(gb + j * 8), AS3V(lb + j * 8), 0, 0);
#else
        v4u ta[4], tb[4];
#pragma unroll
        for (int j = 0; j < 4; ++j) { ta[j] = *(const v4u*)(ga + j * 8); tb[j] = *(const v4u*)(gb + j * 8); }
#pragma unroll
        for (int j = 0; j < 4; ++j) { *(v4u*)(la + j * 8) = ta[j]; *(v4u*)(lb + j * 8) = tb[j]; }
#endif
    };

    stage(0, 0);

    for (int ks = 0; ks < nks; ++ks) {
        const int buf = ks & 1;
        if (ks + 1 < nks) {
            stage(buf ^ 1, (ks + 1) << 6);
#if HAS_ASYNC_LDS
            WAIT_ASYNC(8);              // this buffer done; next stage (8 ops) in flight
#endif
        } else {
#if HAS_ASYNC_LDS
            WAIT_ASYNC(0);
#endif
        }
        __syncthreads();

        const u16* baseA = &sA[buf][((wave & 1) * 64 + ln) * 64];
        const u16* baseB = &sB[buf][((wave >> 1) * 32 + ln) * 64];
#pragma unroll
        for (int kk = 0; kk < 64; kk += 32) {
            Frag fa[4], fb[2];
#pragma unroll
            for (int rt = 0; rt < 4; ++rt) {
                const u16* p = baseA + rt * 16 * 64 + kk + g * 8;
                fa[rt].q[0] = *(const v4u*)p;
                fa[rt].q[1] = *(const v4u*)(p + 16);
            }
#pragma unroll
            for (int ct = 0; ct < 2; ++ct) {
                const u16* p = baseB + ct * 16 * 64 + kk + g * 8;
                fb[ct].q[0] = *(const v4u*)p;
                fb[ct].q[1] = *(const v4u*)(p + 16);
            }
#pragma unroll
            for (int rt = 0; rt < 4; ++rt)
#pragma unroll
                for (int ct = 0; ct < 2; ++ct)
                    acc[rt][ct] = __builtin_amdgcn_wmma_f32_16x16x32_bf16(
                        false, fa[rt].v, false, fb[ct].v, (short)0, acc[rt][ct], false, false);
        }
        __syncthreads();
    }

#pragma unroll
    for (int rt = 0; rt < 4; ++rt)
#pragma unroll
        for (int ct = 0; ct < 2; ++ct)
#pragma unroll
            for (int r = 0; r < 8; ++r) {
                int row = mBase + (wave & 1) * 64 + rt * 16 + r + 8 * g;
                int col = nBase + (wave >> 1) * 32 + ct * 16 + ln;
                size_t idx = (size_t)row * N + col;
                float v = acc[rt][ct][r];
                if (mode == 0)      Cf[idx] = v;
                else if (mode == 1) Cf[idx] = v + R[idx];
                else if (mode == 2) Cb[idx] = f2bf(v);
                else {              // fused SwiGLU: silu(h1) * h3
                    float h = bf2f(Hb[idx]);
                    Cb[idx] = f2bf(h * (1.0f / (1.0f + expf(-h))) * v);
                }
            }
}

// ---------------------------------------------------------------- RoPE + head split + V transpose
// Qx/Kx/Vx: bf16 [B*S][1024];  Qb/Kb: bf16 [B][H][S][64];  Vt: bf16 [B][H][64][S]
__global__ void rope_kernel(const u16* __restrict__ Qx, const u16* __restrict__ Kx,
                            const u16* __restrict__ Vx,
                            u16* __restrict__ Qb, u16* __restrict__ Kb, u16* __restrict__ Vt) {
    int idx = blockIdx.x * 256 + threadIdx.x;          // B*S*H*32 threads
    int i = idx & 31; int t = idx >> 5;
    int h = t & 15;   t >>= 4;
    int s = t & (SEQ - 1); int b = t >> 11;

    float ang = (float)s * powf(10000.0f, -(float)i * (1.0f / 32.0f));
    float sn, cs; sincosf(ang, &sn, &cs);

    size_t rowx = ((size_t)(b * SEQ + s)) * D_MODEL + h * DK;
    size_t rowh = ((size_t)((b * NHEADS + h) * SEQ + s)) * DK;

    float q1 = bf2f(Qx[rowx + 2 * i]), q2 = bf2f(Qx[rowx + 2 * i + 1]);
    Qb[rowh + 2 * i]     = f2bf(q1 * cs - q2 * sn);
    Qb[rowh + 2 * i + 1] = f2bf(q1 * sn + q2 * cs);

    float k1 = bf2f(Kx[rowx + 2 * i]), k2 = bf2f(Kx[rowx + 2 * i + 1]);
    Kb[rowh + 2 * i]     = f2bf(k1 * cs - k2 * sn);
    Kb[rowh + 2 * i + 1] = f2bf(k1 * sn + k2 * cs);

    size_t vb = (size_t)(b * NHEADS + h) * DK;
    Vt[(vb + 2 * i)     * SEQ + s] = Vx[rowx + 2 * i];       // raw bf16 copy
    Vt[(vb + 2 * i + 1) * SEQ + s] = Vx[rowx + 2 * i + 1];
}

// ---------------------------------------------------------------- flash attention (causal), 1 wave / 16-row q tile
__global__ __launch_bounds__(32) void attn_kernel(const u16* __restrict__ Qb,
                                                  const u16* __restrict__ Kb,
                                                  const u16* __restrict__ Vt,
                                                  u16* __restrict__ Ctx) {
    int bid = blockIdx.x;                 // b*H*(S/16)
    int qt = bid & 127; int h = (bid >> 7) & 15; int b = bid >> 11;
    int lane = threadIdx.x, g = lane >> 4, ln = lane & 15;
    int q0 = qt * 16;

    const u16* Qh = Qb + (size_t)(b * NHEADS + h) * SEQ * DK;
    const u16* Kh = Kb + (size_t)(b * NHEADS + h) * SEQ * DK;
    const u16* Vh = Vt + (size_t)(b * NHEADS + h) * DK * SEQ;

    Frag aq0, aq1;
    {
        const u16* p = Qh + (size_t)(q0 + ln) * DK + g * 8;
        aq0.q[0] = *(const v4u*)p;        aq0.q[1] = *(const v4u*)(p + 16);
        aq1.q[0] = *(const v4u*)(p + 32); aq1.q[1] = *(const v4u*)(p + 48);
    }

    float m_[8], l_[8];
    v8f o[4];
#pragma unroll
    for (int r = 0; r < 8; ++r) { m_[r] = -3.0e38f; l_[r] = 0.0f; }
#pragma unroll
    for (int t = 0; t < 4; ++t) o[t] = vzero8();

    __shared__ __attribute__((aligned(32))) u16 Pl[16 * 32];

    int nsteps = (q0 + 16 + 31) >> 5;
    for (int ks = 0; ks < nsteps; ++ks) {
        int key0 = ks * 32;
        v8f s0 = vzero8(), s1 = vzero8();

        {   // scores: keys key0..key0+15
            const u16* pk = Kh + (size_t)(key0 + ln) * DK + g * 8;
            Frag b0, b1;
            b0.q[0] = *(const v4u*)pk;        b0.q[1] = *(const v4u*)(pk + 16);
            b1.q[0] = *(const v4u*)(pk + 32); b1.q[1] = *(const v4u*)(pk + 48);
            s0 = __builtin_amdgcn_wmma_f32_16x16x32_bf16(false, aq0.v, false, b0.v, (short)0, s0, false, false);
            s0 = __builtin_amdgcn_wmma_f32_16x16x32_bf16(false, aq1.v, false, b1.v, (short)0, s0, false, false);
        }
        {   // scores: keys key0+16..key0+31
            const u16* pk = Kh + (size_t)(key0 + 16 + ln) * DK + g * 8;
            Frag b0, b1;
            b0.q[0] = *(const v4u*)pk;        b0.q[1] = *(const v4u*)(pk + 16);
            b1.q[0] = *(const v4u*)(pk + 32); b1.q[1] = *(const v4u*)(pk + 48);
            s1 = __builtin_amdgcn_wmma_f32_16x16x32_bf16(false, aq0.v, false, b0.v, (short)0, s1, false, false);
            s1 = __builtin_amdgcn_wmma_f32_16x16x32_bf16(false, aq1.v, false, b1.v, (short)0, s1, false, false);
        }

        float p0[8], p1[8];
#pragma unroll
        for (int r = 0; r < 8; ++r) {
            int q = q0 + r + 8 * g;
            float v0 = s0[r] * 0.125f, v1 = s1[r] * 0.125f;
            v0 = (key0 + ln      <= q) ? v0 : -1.0e30f;
            v1 = (key0 + 16 + ln <= q) ? v1 : -1.0e30f;
            float t = fmaxf(v0, v1);
            for (int msk = 1; msk < 16; msk <<= 1) t = fmaxf(t, __shfl_xor(t, msk, 32));
            float mnew = fmaxf(m_[r], t);
            float alpha = expf(m_[r] - mnew);
            p0[r] = expf(v0 - mnew);
            p1[r] = expf(v1 - mnew);
            float rs = p0[r] + p1[r];
            for (int msk = 1; msk < 16; msk <<= 1) rs += __shfl_xor(rs, msk, 32);
            l_[r] = l_[r] * alpha + rs;
            m_[r] = mnew;
#pragma unroll
            for (int t4 = 0; t4 < 4; ++t4) o[t4][r] *= alpha;
        }

        // P (C layout) -> LDS row-major 16x32 bf16
#pragma unroll
        for (int r = 0; r < 8; ++r) {
            int row = r + 8 * g;
            Pl[row * 32 + ln]      = f2bf(p0[r]);
            Pl[row * 32 + 16 + ln] = f2bf(p1[r]);
        }
        __syncthreads();
        Frag ap;
        {
            const u16* p = &Pl[ln * 32 + g * 8];
            ap.q[0] = *(const v4u*)p;
            ap.q[1] = *(const v4u*)(p + 16);
        }
        __syncthreads();

        // o += P (16x32) @ V (32x64)
#pragma unroll
        for (int t4 = 0; t4 < 4; ++t4) {
            const u16* pv = Vh + (size_t)(t4 * 16 + ln) * SEQ + key0 + g * 8;
            Frag bv;
            bv.q[0] = *(const v4u*)pv;
            bv.q[1] = *(const v4u*)(pv + 16);
            o[t4] = __builtin_amdgcn_wmma_f32_16x16x32_bf16(false, ap.v, false, bv.v, (short)0, o[t4], false, false);
        }
    }

    // normalize + write ctx (bf16, [B*S][1024])
#pragma unroll
    for (int r = 0; r < 8; ++r) {
        float inv = 1.0f / l_[r];
        int row = b * SEQ + q0 + r + 8 * g;
#pragma unroll
        for (int t4 = 0; t4 < 4; ++t4) {
            int col = h * DK + t4 * 16 + ln;
            Ctx[(size_t)row * D_MODEL + col] = f2bf(o[t4][r] * inv);
        }
    }
}

// ---------------------------------------------------------------- launch
extern "C" void kernel_launch(void* const* d_in, const int* in_sizes, int n_in,
                              void* d_out, int out_size, void* d_ws, size_t ws_size,
                              hipStream_t stream) {
    const float* X  = (const float*)d_in[0];
    const float* WQ = (const float*)d_in[1];
    const float* WK = (const float*)d_in[2];
    const float* WV = (const float*)d_in[3];
    const float* WO = (const float*)d_in[4];
    const float* W1 = (const float*)d_in[5];
    const float* W2 = (const float*)d_in[6];
    const float* W3 = (const float*)d_in[7];
    const float* g1 = (const float*)d_in[8];
    const float* g2 = (const float*)d_in[9];
    float* out = (float*)d_out;

    char* ws = (char*)d_ws;
    size_t cur = 0;
    auto alloc = [&](size_t bytes) { char* p = ws + cur; cur += (bytes + 255) & ~size_t(255); return p; };

    u16* Wbf = (u16*)alloc(16777216ull * 2);   // all 7 weights, bf16
    u16* wq = Wbf + 0,       *wk = Wbf + 1048576, *wv = Wbf + 2097152, *wo = Wbf + 3145728;
    u16* w1 = Wbf + 4194304, *w2 = Wbf + 8388608, *w3 = Wbf + 12582912;

    u16* Xn  = (u16*)alloc((size_t)MROWS * D_MODEL * 2);
    u16* Qx  = (u16*)alloc((size_t)MROWS * D_MODEL * 2);
    u16* Kx  = (u16*)alloc((size_t)MROWS * D_MODEL * 2);
    u16* Vx  = (u16*)alloc((size_t)MROWS * D_MODEL * 2);
    u16* Qb  = (u16*)alloc((size_t)MROWS * D_MODEL * 2);
    u16* Kb  = (u16*)alloc((size_t)MROWS * D_MODEL * 2);
    u16* Vt  = (u16*)alloc((size_t)MROWS * D_MODEL * 2);
    u16* Ctx = (u16*)alloc((size_t)MROWS * D_MODEL * 2);
    float* Y = (float*)alloc((size_t)MROWS * D_MODEL * 4);
    u16* Yn  = (u16*)alloc((size_t)MROWS * D_MODEL * 2);
    u16* H1b = (u16*)alloc((size_t)MROWS * D_FF * 2);
    u16* FFb = (u16*)alloc((size_t)MROWS * D_FF * 2);
    (void)ws_size; (void)in_sizes; (void)n_in; (void)out_size;

    // 1) weights -> bf16
    f2bf_kernel<<<dim3(4096),  dim3(256), 0, stream>>>(WQ, wq, 1048576);
    f2bf_kernel<<<dim3(4096),  dim3(256), 0, stream>>>(WK, wk, 1048576);
    f2bf_kernel<<<dim3(4096),  dim3(256), 0, stream>>>(WV, wv, 1048576);
    f2bf_kernel<<<dim3(4096),  dim3(256), 0, stream>>>(WO, wo, 1048576);
    f2bf_kernel<<<dim3(16384), dim3(256), 0, stream>>>(W1, w1, 4194304);
    f2bf_kernel<<<dim3(16384), dim3(256), 0, stream>>>(W2, w2, 4194304);
    f2bf_kernel<<<dim3(16384), dim3(256), 0, stream>>>(W3, w3, 4194304);

    // 2) Xn = rmsnorm(X, g1) in bf16
    rmsnorm_kernel<<<dim3(MROWS), dim3(256), 0, stream>>>(X, g1, Xn);

    // 3) Q/K/V projections (bf16 out)
    dim3 blk(256);
    dim3 grdD(D_MODEL / 128, MROWS / 128);
    gemm_kernel<<<grdD, blk, 0, stream>>>(Xn, wq, D_MODEL, D_MODEL, 2, nullptr, nullptr, Qx, nullptr);
    gemm_kernel<<<grdD, blk, 0, stream>>>(Xn, wk, D_MODEL, D_MODEL, 2, nullptr, nullptr, Kx, nullptr);
    gemm_kernel<<<grdD, blk, 0, stream>>>(Xn, wv, D_MODEL, D_MODEL, 2, nullptr, nullptr, Vx, nullptr);

    // 4) RoPE + head split + V transpose
    rope_kernel<<<dim3(16384), dim3(256), 0, stream>>>(Qx, Kx, Vx, Qb, Kb, Vt);

    // 5) causal flash attention -> Ctx (bf16)
    attn_kernel<<<dim3(BATCH * NHEADS * (SEQ / 16)), dim3(32), 0, stream>>>(Qb, Kb, Vt, Ctx);

    // 6) Y = X + Ctx @ WO^T   (fp32)
    gemm_kernel<<<grdD, blk, 0, stream>>>(Ctx, wo, D_MODEL, D_MODEL, 1, Y, X, nullptr, nullptr);

    // 7) Yn = rmsnorm(Y, g2)
    rmsnorm_kernel<<<dim3(MROWS), dim3(256), 0, stream>>>(Y, g2, Yn);

    // 8) h1 = Yn @ W1^T (bf16)
    dim3 grdF(D_FF / 128, MROWS / 128);
    gemm_kernel<<<grdF, blk, 0, stream>>>(Yn, w1, D_FF, D_MODEL, 2, nullptr, nullptr, H1b, nullptr);

    // 9) ff_in = silu(h1) * (Yn @ W3^T)   (fused SwiGLU epilogue, bf16)
    gemm_kernel<<<grdF, blk, 0, stream>>>(Yn, w3, D_FF, D_MODEL, 3, nullptr, nullptr, FFb, H1b);

    // 10) out = Y + ff_in @ W2^T   (fp32)
    gemm_kernel<<<grdD, blk, 0, stream>>>(FFb, w2, D_MODEL, D_FF, 1, out, Y, nullptr, nullptr);
}
